// MessagePassingJax_51874615001132
// MI455X (gfx1250) — matrix operand order repaired and checked by reading, not compiled, not run on hardware
//
#include <hip/hip_runtime.h>
#include <stdint.h>
#include <stddef.h>

// ---------------------------------------------------------------------------
// GNN message passing for MI455X (gfx1250), wave32 + V_WMMA_F32_16X16X32_BF16.
//   edge MLP (bf16 WMMA, f32 accum) -> f32 atomic segment-sum -> node MLP.
// Weights are pre-swizzled once per launch into the WMMA B-fragment layout
// (bf16), so inner-loop B loads are 2x global_load_b128 instead of 16x b32.
// ---------------------------------------------------------------------------

typedef __bf16 bf16_t;
typedef __attribute__((ext_vector_type(16))) __bf16 v16bf;
typedef __attribute__((ext_vector_type(8)))  __bf16 v8bf;
typedef __attribute__((ext_vector_type(8)))  float  v8f;

#define GNN_D    128
#define GNN_2D   256
#define ROWS_PB  32     // rows (edges/nodes) per block = 2 row-tiles of 16
#define NTHREADS 256    // 8 wave32 waves; wave w owns output cols [16w,16w+16)

union V16 {
  v16bf v;
  struct Halves { v8bf lo; v8bf hi; } h;
  bf16_t e[16];
};

// A-fragment (16x32 bf16) for one 32-wide K chunk, in the contiguous k order also used by the packed B:
// lane half 'hs' (lane>>4): elems 0..15 = K k0+16*hs .. k0+16*hs+15 (contiguous)
__device__ __forceinline__ v16bf load_a_frag(const bf16_t* row, int k0, int hs) {
  V16 u;
  u.h.lo = *(const v8bf*)(row + k0 + hs * 16);
  u.h.hi = *(const v8bf*)(row + k0 + hs * 16 + 8);
  return u.v;
}

// B-fragment from pre-swizzled weights: fragment (kc, wv) stored as
// 32 lanes x 16 bf16 contiguous (32 B per lane) -> two global_load_b128.
__device__ __forceinline__ v16bf load_b_frag_packed(const bf16_t* __restrict__ pw,
                                                    int kc, int wv, int lane) {
  const bf16_t* p = pw + (size_t)(((kc * 8 + wv) * 32 + lane) * 16);
  V16 u;
  u.h.lo = *(const v8bf*)(p);
  u.h.hi = *(const v8bf*)(p + 8);
  return u.v;
}

__global__ __launch_bounds__(NTHREADS) void gnn_zero_f4(float4* __restrict__ p, int n4) {
  int i = blockIdx.x * blockDim.x + threadIdx.x;
  if (i < n4) p[i] = make_float4(0.f, 0.f, 0.f, 0.f);
}

// ---------------------------------------------------------------------------
// Weight pre-swizzle: W is [nkc*32, 128] f32 row-major. One thread per
// (kc, wv, lane) writes that lane's 16 contiguous bf16:
//   element j = W[(kc*32 + (lane>>4)*16 + j) * 128 + wv*16 + (lane&15)]
// ---------------------------------------------------------------------------
__global__ __launch_bounds__(NTHREADS) void gnn_pack_w(
    const float* __restrict__ W, bf16_t* __restrict__ dst, int nfrag /* = nkc*256 */) {
  const int t = blockIdx.x * blockDim.x + threadIdx.x;
  if (t >= nfrag) return;
  const int kc   = t >> 8;          // t / 256
  const int rem  = t & 255;
  const int wv   = rem >> 5;        // fragment column tile
  const int lane = rem & 31;
  const int col  = wv * 16 + (lane & 15);
  const int krow = kc * 32 + (lane >> 4) * 16;
  bf16_t* o = dst + (size_t)t * 16;
#pragma unroll
  for (int j = 0; j < 16; ++j)
    o[j] = (bf16_t)W[(size_t)(krow + j) * GNN_D + col];
}

// ---------------------------------------------------------------------------
// Edge kernel: X = [node[senders[e]] | edge[e]] (bf16 in LDS)
//   H = relu(X@W1 + b1) ; M = H@W2 + b2 ; atomicAdd into agg[receivers[e]].
// ---------------------------------------------------------------------------
__global__ __launch_bounds__(NTHREADS) void gnn_edge_mlp(
    const float* __restrict__ node_lat, const float* __restrict__ edge_lat,
    const int* __restrict__ senders, const int* __restrict__ receivers,
    const bf16_t* __restrict__ pW1, const float* __restrict__ b1,
    const bf16_t* __restrict__ pW2, const float* __restrict__ b2,
    float* __restrict__ agg, int nEdges)
{
  __shared__ bf16_t sX[ROWS_PB][GNN_2D];  // 16 KB
  __shared__ bf16_t sH[ROWS_PB][GNN_D];   //  8 KB

  const int t  = threadIdx.x;
  const int e0 = blockIdx.x * ROWS_PB;

  // ---- stage X tile (gather + concat), f32 -> bf16 ----
  {
    const int srow = t >> 3;            // 0..31
    const int cb   = (t & 7) * 32;      // 0,32,...,224
    int e = e0 + srow; if (e >= nEdges) e = nEdges - 1;   // clamp tail (masked later)
    const float* src = (cb < GNN_D)
        ? node_lat + (size_t)senders[e] * GNN_D + cb
        : edge_lat + (size_t)e * GNN_D + (cb - GNN_D);
#pragma unroll
    for (int j = 0; j < 32; j += 4) {
      const float4 v = *(const float4*)(src + j);
      sX[srow][cb + j + 0] = (bf16_t)v.x;
      sX[srow][cb + j + 1] = (bf16_t)v.y;
      sX[srow][cb + j + 2] = (bf16_t)v.z;
      sX[srow][cb + j + 3] = (bf16_t)v.w;
    }
  }
  __syncthreads();

  const int lane = t & 31, wv = t >> 5;
  const int hs   = lane >> 4;           // lane half select
  const int lcol = lane & 15;
  const int n0   = wv * 16;             // this wave's output column tile

  v8f acc0, acc1;

  // ---- GEMM1: H = relu(X @ W1 + b1), K = 256 ----
  {
    const float bias = b1[n0 + lcol];
#pragma unroll
    for (int r = 0; r < 8; ++r) { acc0[r] = bias; acc1[r] = bias; }
    const bf16_t* xr0 = sX[lcol];        // A row (M = lane&15), tile 0
    const bf16_t* xr1 = sX[16 + lcol];   // tile 1
#pragma unroll
    for (int kc = 0; kc < GNN_2D / 32; ++kc) {
      const int k0 = kc * 32;
      const v16bf bfr = load_b_frag_packed(pW1, kc, wv, lane);
      acc0 = __builtin_amdgcn_wmma_f32_16x16x32_bf16(
          false, load_a_frag(xr0, k0, hs), false, bfr, (short)0, acc0, false, false);
      acc1 = __builtin_amdgcn_wmma_f32_16x16x32_bf16(
          false, load_a_frag(xr1, k0, hs), false, bfr, (short)0, acc1, false, false);
    }
#pragma unroll
    for (int r = 0; r < 8; ++r) {        // C layout: row = r + 8*hs (+16 for tile 1)
      float h0 = acc0[r] > 0.f ? acc0[r] : 0.f;
      float h1 = acc1[r] > 0.f ? acc1[r] : 0.f;
      sH[hs * 8 + r][n0 + lcol]      = (bf16_t)h0;
      sH[16 + hs * 8 + r][n0 + lcol] = (bf16_t)h1;
    }
  }
  __syncthreads();

  // ---- GEMM2: M = H @ W2 + b2, K = 128 ----
  {
    const float bias = b2[n0 + lcol];
#pragma unroll
    for (int r = 0; r < 8; ++r) { acc0[r] = bias; acc1[r] = bias; }
    const bf16_t* hr0 = sH[lcol];
    const bf16_t* hr1 = sH[16 + lcol];
#pragma unroll
    for (int kc = 0; kc < GNN_D / 32; ++kc) {
      const int k0 = kc * 32;
      const v16bf bfr = load_b_frag_packed(pW2, kc, wv, lane);
      acc0 = __builtin_amdgcn_wmma_f32_16x16x32_bf16(
          false, load_a_frag(hr0, k0, hs), false, bfr, (short)0, acc0, false, false);
      acc1 = __builtin_amdgcn_wmma_f32_16x16x32_bf16(
          false, load_a_frag(hr1, k0, hs), false, bfr, (short)0, acc1, false, false);
    }
  }

  // ---- segment_sum: f32 atomics into agg[receiver] (L2-resident, 25.6 MB) ----
#pragma unroll
  for (int r = 0; r < 8; ++r) {
    const int ra = hs * 8 + r, rb = 16 + hs * 8 + r;
    const int ea = e0 + ra,   ebb = e0 + rb;
    if (ea  < nEdges) atomicAdd(&agg[(size_t)receivers[ea]  * GNN_D + n0 + lcol], acc0[r]);
    if (ebb < nEdges) atomicAdd(&agg[(size_t)receivers[ebb] * GNN_D + n0 + lcol], acc1[r]);
  }
}

// ---------------------------------------------------------------------------
// Node kernel: X = [node[i] | agg[i]] ; out[i] = relu(X@U1+ub1)@U2 + ub2.
// ---------------------------------------------------------------------------
__global__ __launch_bounds__(NTHREADS) void gnn_node_mlp(
    const float* __restrict__ node_lat, const float* __restrict__ agg,
    const bf16_t* __restrict__ pU1, const float* __restrict__ ub1,
    const bf16_t* __restrict__ pU2, const float* __restrict__ ub2,
    float* __restrict__ out, int nNodes)
{
  __shared__ bf16_t sX[ROWS_PB][GNN_2D];
  __shared__ bf16_t sH[ROWS_PB][GNN_D];

  const int t  = threadIdx.x;
  const int i0 = blockIdx.x * ROWS_PB;

  {
    const int srow = t >> 3;
    const int cb   = (t & 7) * 32;
    int i = i0 + srow; if (i >= nNodes) i = nNodes - 1;
    const float* src = (cb < GNN_D)
        ? node_lat + (size_t)i * GNN_D + cb
        : agg      + (size_t)i * GNN_D + (cb - GNN_D);
#pragma unroll
    for (int j = 0; j < 32; j += 4) {
      const float4 v = *(const float4*)(src + j);
      sX[srow][cb + j + 0] = (bf16_t)v.x;
      sX[srow][cb + j + 1] = (bf16_t)v.y;
      sX[srow][cb + j + 2] = (bf16_t)v.z;
      sX[srow][cb + j + 3] = (bf16_t)v.w;
    }
  }
  __syncthreads();

  const int lane = t & 31, wv = t >> 5;
  const int hs   = lane >> 4;
  const int lcol = lane & 15;
  const int n0   = wv * 16;

  v8f acc0, acc1;

  {
    const float bias = ub1[n0 + lcol];
#pragma unroll
    for (int r = 0; r < 8; ++r) { acc0[r] = bias; acc1[r] = bias; }
    const bf16_t* xr0 = sX[lcol];
    const bf16_t* xr1 = sX[16 + lcol];
#pragma unroll
    for (int kc = 0; kc < GNN_2D / 32; ++kc) {
      const int k0 = kc * 32;
      const v16bf bfr = load_b_frag_packed(pU1, kc, wv, lane);
      acc0 = __builtin_amdgcn_wmma_f32_16x16x32_bf16(
          false, load_a_frag(xr0, k0, hs), false, bfr, (short)0, acc0, false, false);
      acc1 = __builtin_amdgcn_wmma_f32_16x16x32_bf16(
          false, load_a_frag(xr1, k0, hs), false, bfr, (short)0, acc1, false, false);
    }
#pragma unroll
    for (int r = 0; r < 8; ++r) {
      float h0 = acc0[r] > 0.f ? acc0[r] : 0.f;
      float h1 = acc1[r] > 0.f ? acc1[r] : 0.f;
      sH[hs * 8 + r][n0 + lcol]      = (bf16_t)h0;
      sH[16 + hs * 8 + r][n0 + lcol] = (bf16_t)h1;
    }
  }
  __syncthreads();

  {
    const float bias = ub2[n0 + lcol];
#pragma unroll
    for (int r = 0; r < 8; ++r) { acc0[r] = bias; acc1[r] = bias; }
    const bf16_t* hr0 = sH[lcol];
    const bf16_t* hr1 = sH[16 + lcol];
#pragma unroll
    for (int kc = 0; kc < GNN_D / 32; ++kc) {
      const int k0 = kc * 32;
      const v16bf bfr = load_b_frag_packed(pU2, kc, wv, lane);
      acc0 = __builtin_amdgcn_wmma_f32_16x16x32_bf16(
          false, load_a_frag(hr0, k0, hs), false, bfr, (short)0, acc0, false, false);
      acc1 = __builtin_amdgcn_wmma_f32_16x16x32_bf16(
          false, load_a_frag(hr1, k0, hs), false, bfr, (short)0, acc1, false, false);
    }
  }

#pragma unroll
  for (int r = 0; r < 8; ++r) {
    const int ia = i0 + hs * 8 + r;
    const int ib = i0 + 16 + hs * 8 + r;
    if (ia < nNodes) out[(size_t)ia * GNN_D + n0 + lcol] = acc0[r];
    if (ib < nNodes) out[(size_t)ib * GNN_D + n0 + lcol] = acc1[r];
  }
}

// ---------------------------------------------------------------------------
extern "C" void kernel_launch(void* const* d_in, const int* in_sizes, int n_in,
                              void* d_out, int out_size, void* d_ws, size_t ws_size,
                              hipStream_t stream) {
  (void)n_in; (void)out_size; (void)ws_size;
  const float* node_lat = (const float*)d_in[0];
  const float* edge_lat = (const float*)d_in[1];
  const int*   eidx     = (const int*)d_in[2];
  const float* W1  = (const float*)d_in[3];
  const float* b1  = (const float*)d_in[4];
  const float* W2  = (const float*)d_in[5];
  const float* b2  = (const float*)d_in[6];
  const float* U1  = (const float*)d_in[7];
  const float* ub1 = (const float*)d_in[8];
  const float* U2  = (const float*)d_in[9];
  const float* ub2 = (const float*)d_in[10];

  const int N = in_sizes[0] / GNN_D;    // 50000
  const int E = in_sizes[1] / GNN_D;    // 625000
  const int* senders   = eidx;          // edge_index[0]
  const int* receivers = eidx + E;      // edge_index[1]

  // workspace layout: agg [N*D f32] | pW1 | pW2 | pU1 | pU2 (bf16, 32B aligned)
  float*  agg = (float*)d_ws;
  bf16_t* pW1 = (bf16_t*)(agg + (size_t)N * GNN_D);
  bf16_t* pW2 = pW1 + (size_t)GNN_2D * GNN_D;   // 32768 elems
  bf16_t* pU1 = pW2 + (size_t)GNN_D  * GNN_D;   // 16384 elems
  bf16_t* pU2 = pU1 + (size_t)GNN_2D * GNN_D;

  float* out_nodes = (float*)d_out;                      // [N, D]
  float* out_edges = out_nodes + (size_t)N * GNN_D;      // [E, D] pass-through

  // 0) pre-swizzle weights into WMMA B-fragment layout (bf16)
  hipLaunchKernelGGL(gnn_pack_w, dim3(8), dim3(NTHREADS), 0, stream, W1, pW1, 8 * 256);
  hipLaunchKernelGGL(gnn_pack_w, dim3(4), dim3(NTHREADS), 0, stream, W2, pW2, 4 * 256);
  hipLaunchKernelGGL(gnn_pack_w, dim3(8), dim3(NTHREADS), 0, stream, U1, pU1, 8 * 256);
  hipLaunchKernelGGL(gnn_pack_w, dim3(4), dim3(NTHREADS), 0, stream, U2, pU2, 4 * 256);

  // 1) zero aggregation buffer (ws is stale/poisoned between replays)
  const int n4 = (N * GNN_D) / 4;
  hipLaunchKernelGGL(gnn_zero_f4, dim3((n4 + NTHREADS - 1) / NTHREADS),
                     dim3(NTHREADS), 0, stream, (float4*)agg, n4);

  // 2) edge MLP + atomic segment-sum
  const int ebk = (E + ROWS_PB - 1) / ROWS_PB;
  hipLaunchKernelGGL(gnn_edge_mlp, dim3(ebk), dim3(NTHREADS), 0, stream,
                     node_lat, edge_lat, senders, receivers, pW1, b1, pW2, b2, agg, E);

  // 3) node MLP
  const int nbk = (N + ROWS_PB - 1) / ROWS_PB;
  hipLaunchKernelGGL(gnn_node_mlp, dim3(nbk), dim3(NTHREADS), 0, stream,
                     node_lat, agg, pU1, ub1, pU2, ub2, out_nodes, N);

  // 4) second output = unchanged edge_latents
  hipMemcpyAsync(out_edges, edge_lat, (size_t)E * GNN_D * sizeof(float),
                 hipMemcpyDeviceToDevice, stream);
}
